// stackedIndRNN_encoder_7318624272385
// MI455X (gfx1250) — matrix operand order repaired and checked by reading, not compiled, not run on hardware
//
#include <hip/hip_runtime.h>

// Problem constants (match reference)
#define Tt 100
#define Bb 128
#define Hh 2048
#define Ll 6
#define Vv 50
#define EPSf 1e-5f
#define KT 64          // Hh / 32 k-tiles
#define NT 128         // Hh / 16 n-tiles

#define DEPTH     4        // async pipeline stages
#define STAGE_B   5120     // bytes per stage per wave: A 1KB + 4x B 1KB
#define WAVE_B    (DEPTH * STAGE_B)       // 20480 B per wave
#define LDS_DYN   (8 * WAVE_B)            // 163840 B per block

typedef __attribute__((ext_vector_type(16))) __bf16       v16bf;
typedef __attribute__((ext_vector_type(8)))  float        v8f;
typedef __attribute__((ext_vector_type(8)))  unsigned int v8u;

typedef __attribute__((address_space(3))) const char* lds_cptr;
__device__ __forceinline__ unsigned lds_off(const void* p) {
  return (unsigned)(unsigned long long)(lds_cptr)p;
}

// one async b128: each lane copies 16B global -> 16B LDS (512B per wave-op)
#define ASYNC_CP16(ldsbyte, gptr)                                          \
  asm volatile("global_load_async_to_lds_b128 %0, %1, off"                 \
               :: "v"(ldsbyte), "v"(gptr) : "memory")

#define WAIT_ASYNC(n)                                                      \
  asm volatile("s_wait_asynccnt " #n ::: "memory")

// float -> bf16 bits, round-to-nearest-even
__device__ __forceinline__ unsigned short f2bf(float f) {
  unsigned int u = __builtin_bit_cast(unsigned int, f);
  u += 0x7fffu + ((u >> 16) & 1u);
  return (unsigned short)(u >> 16);
}

// Packed A-fragment index for element (m, k), K-dim = Hh.
__device__ __forceinline__ int a_pack_idx(int m, int k) {
  int mt  = m >> 4;
  int kt  = k >> 5;
  int kin = k & 31;
  int hi  = ((kin >= 8) && (kin < 16)) || (kin >= 24);
  int lane = (m & 15) + (hi ? 16 : 0);
  int e    = (kin & 7) + ((kin >= 16) ? 8 : 0);
  return ((mt * KT) + kt) * 512 + lane * 16 + e;
}

// ---------------------------------------------------------------------------
// One-time (per launch) weight repack: fp32 [L,H,H] -> bf16 B-fragment order.
__global__ void pack_weights(const float* __restrict__ Ws,
                             unsigned short* __restrict__ Wp) {
  const long total = (long)Ll * Hh * Hh;
  for (long i = (long)blockIdx.x * blockDim.x + threadIdx.x; i < total;
       i += (long)gridDim.x * blockDim.x) {
    int  e    = (int)(i & 15);
    int  lane = (int)((i >> 4) & 31);
    long rest = i >> 9;
    int  kt   = (int)(rest & (KT - 1));
    rest >>= 6;
    int  nt   = (int)(rest & (NT - 1));
    int  l    = (int)(rest >> 7);
    int  n = nt * 16 + (lane & 15);
    int  k = kt * 32 + ((lane >> 4) << 3) + (e & 7) + ((e >> 3) << 4);
    Wp[i] = f2bf(Ws[((long)l * Hh + n) * Hh + k]);
  }
}

// Wout [V,H] -> bf16 fragments, N padded 50 -> 64 with zeros.
__global__ void pack_wout(const float* __restrict__ Wout,
                          unsigned short* __restrict__ WoutP) {
  const int total = 64 * Hh;
  for (int i = blockIdx.x * blockDim.x + threadIdx.x; i < total;
       i += gridDim.x * blockDim.x) {
    int e    = i & 15;
    int lane = (i >> 4) & 31;
    int rest = i >> 9;
    int kt   = rest & (KT - 1);
    int nt   = rest >> 6;
    int n = nt * 16 + (lane & 15);
    int k = kt * 32 + ((lane >> 4) << 3) + (e & 7) + ((e >> 3) << 4);
    WoutP[i] = (n < Vv) ? f2bf(Wout[n * Hh + k]) : (unsigned short)0;
  }
}

// Embedding gather for one timestep -> packed A-fragment bf16 [B x H].
__global__ void embed_gather(const int* __restrict__ tok,
                             const float* __restrict__ embed,
                             unsigned short* __restrict__ xA) {
  int i = blockIdx.x * blockDim.x + threadIdx.x;   // over B*H
  int b = i >> 11;
  int k = i & (Hh - 1);
  xA[a_pack_idx(b, k)] = f2bf(embed[(long)tok[b] * Hh + k]);
}

// ---------------------------------------------------------------------------
// Fused per-(t,l) kernel:  s = x @ W^T  (bf16 WMMA, fp32 acc)
//   + batch-norm over the 128 batch rows (block-local)
//   + h' = relu(s_bn*gamma + beta + u*h), dual-format writeback.
// grid = 32 blocks (64 output cols), block = 256 threads = 8 waves.
// Global->LDS async copies (ASYNCcnt), depth-4 wave-private pipeline: the
// compiler cannot collapse it, so WMMAs run ~3 L2-latencies behind the loads.
__global__ void __launch_bounds__(256) layer_step(
    const unsigned short* __restrict__ xA,     // packed A  [8 mt][64 kt][512]
    const unsigned short* __restrict__ Wp,     // packed B  [128 nt][64 kt][512]
    const float* __restrict__ gamma,
    const float* __restrict__ beta,
    const float* __restrict__ u,
    float* __restrict__ h,                     // [B,H] state, updated in place
    unsigned short* __restrict__ xOut) {       // packed A for next layer
  extern __shared__ char smem[];               // LDS_DYN bytes
  const int w    = threadIdx.x >> 5;           // wave id: m-tile
  const int lane = threadIdx.x & 31;
  const int nb   = blockIdx.x;                 // 64-column slab

  __shared__ float s1[64];
  __shared__ float s2[64];
  if (threadIdx.x < 64) { s1[threadIdx.x] = 0.f; s2[threadIdx.x] = 0.f; }
  __syncthreads();

  char* reg = smem + w * WAVE_B;                       // this wave's region
  const unsigned short* aT = xA + ((w * KT) << 9);     // A tiles, this m-tile
  const unsigned short* bT = Wp + (((nb * 4) * KT) << 9);

  // stage issue: 2 async b128 for A tile, 2 for each of 4 B tiles (10 total)
#define ISSUE_STAGE(q, slot)                                               \
  {                                                                        \
    char* st = reg + (slot) * STAGE_B;                                     \
    const unsigned short* ga = aT + ((q) << 9);                            \
    ASYNC_CP16(lds_off(st + lane * 16),       ga + lane * 8);              \
    ASYNC_CP16(lds_off(st + 512 + lane * 16), ga + 256 + lane * 8);        \
    _Pragma("unroll")                                                      \
    for (int j = 0; j < 4; ++j) {                                          \
      const unsigned short* gb = bT + (((j * KT) + (q)) << 9);             \
      char* sb = st + 1024 + j * 1024;                                     \
      ASYNC_CP16(lds_off(sb + lane * 16),       gb + lane * 8);            \
      ASYNC_CP16(lds_off(sb + 512 + lane * 16), gb + 256 + lane * 8);      \
    }                                                                      \
  }

#define COMPUTE_STAGE(slot)                                                \
  {                                                                        \
    const char* st = reg + (slot) * STAGE_B;                               \
    v8u av = *(const v8u*)(st + lane * 32);                                \
    v16bf a = __builtin_bit_cast(v16bf, av);                               \
    _Pragma("unroll")                                                      \
    for (int j = 0; j < 4; ++j) {                                          \
      v8u bv = *(const v8u*)(st + 1024 + j * 1024 + lane * 32);            \
      v16bf b = __builtin_bit_cast(v16bf, bv);                             \
      acc[j] = __builtin_amdgcn_wmma_f32_16x16x32_bf16(                    \
          false, a, false, b, (short)0, acc[j], false, false);             \
    }                                                                      \
  }

  v8f acc[4];
#pragma unroll
  for (int j = 0; j < 4; ++j) acc[j] = (v8f){0.f,0.f,0.f,0.f,0.f,0.f,0.f,0.f};

  // prologue: 3 stages in flight
  ISSUE_STAGE(0, 0)
  ISSUE_STAGE(1, 1)
  ISSUE_STAGE(2, 2)

#pragma unroll
  for (int kt = 0; kt < KT - 3; ++kt) {
    ISSUE_STAGE(kt + 3, (kt + 3) & 3)
    WAIT_ASYNC(30);                       // oldest stage resident (in-order)
    COMPUTE_STAGE(kt & 3)
  }
  WAIT_ASYNC(20); COMPUTE_STAGE((KT - 3) & 3)
  WAIT_ASYNC(10); COMPUTE_STAGE((KT - 2) & 3)
  WAIT_ASYNC(0);  COMPUTE_STAGE((KT - 1) & 3)

#undef ISSUE_STAGE
#undef COMPUTE_STAGE

  // Batch-norm statistics over M=128 rows (all resident in this block).
#pragma unroll
  for (int j = 0; j < 4; ++j) {
    float p1 = 0.f, p2 = 0.f;
#pragma unroll
    for (int r = 0; r < 8; ++r) { float v = acc[j][r]; p1 += v; p2 += v * v; }
    const int col = j * 16 + (lane & 15);
    atomicAdd(&s1[col], p1);
    atomicAdd(&s2[col], p2);
  }
  __syncthreads();

  // Epilogue: BN + IndRNN recurrence + ReLU, dual-format writeback.
#pragma unroll
  for (int j = 0; j < 4; ++j) {
    const int col = j * 16 + (lane & 15);
    const int n   = nb * 64 + col;
    const float mu   = s1[col] * (1.f / Bb);
    const float var  = s2[col] * (1.f / Bb) - mu * mu;
    const float rstd = rsqrtf(var + EPSf);
    const float g = gamma[n], be = beta[n], uu = u[n];
#pragma unroll
    for (int r = 0; r < 8; ++r) {
      const int m = (w << 4) + r + ((lane >> 4) << 3);   // C-layout row
      float hv = h[m * Hh + n];
      float v  = (acc[j][r] - mu) * rstd * g + be + uu * hv;
      v = fmaxf(v, 0.f);
      h[m * Hh + n] = v;                  // fp32 recurrent state / final output
      xOut[a_pack_idx(m, n)] = f2bf(v);   // bf16 A-fragment for next GEMM
    }
  }
}

// ---------------------------------------------------------------------------
// Per-timestep output projection: [128 x 2048] @ [2048 x 64(pad)] -> logits.
__global__ void __launch_bounds__(256) out_proj(
    const unsigned short* __restrict__ xA,     // packed A of layer-5 output
    const unsigned short* __restrict__ WoutP,  // packed B [4 nt][64 kt][512]
    const float* __restrict__ bout,
    float* __restrict__ logits) {              // [128 x V] for this timestep
  const int w    = threadIdx.x >> 5;
  const int lane = threadIdx.x & 31;

  const unsigned short* aBase = xA + ((w * KT) << 9) + (lane << 4);
  const unsigned short* bBase = WoutP + (lane << 4);

#define LD_A(kt)    (*(const v8u*)(aBase + ((kt) << 9)))
#define LD_B(j,kt)  (*(const v8u*)(bBase + (((j) * KT + (kt)) << 9)))

  v8f acc[4];
#pragma unroll
  for (int j = 0; j < 4; ++j) acc[j] = (v8f){0.f,0.f,0.f,0.f,0.f,0.f,0.f,0.f};

  v8u aS[3];
  v8u bS[3][4];
#pragma unroll
  for (int s = 0; s < 3; ++s) {
    aS[s] = LD_A(s);
#pragma unroll
    for (int j = 0; j < 4; ++j) bS[s][j] = LD_B(j, s);
  }

#pragma unroll
  for (int kt = 0; kt < KT; ++kt) {
    const int cur = kt % 3;
    v16bf a = __builtin_bit_cast(v16bf, aS[cur]);
#pragma unroll
    for (int j = 0; j < 4; ++j) {
      v16bf b = __builtin_bit_cast(v16bf, bS[cur][j]);
      acc[j] = __builtin_amdgcn_wmma_f32_16x16x32_bf16(
          false, a, false, b, (short)0, acc[j], false, false);
    }
    if (kt + 3 < KT) {
      aS[cur] = LD_A(kt + 3);
#pragma unroll
      for (int j = 0; j < 4; ++j) bS[cur][j] = LD_B(j, kt + 3);
    }
  }
#undef LD_A
#undef LD_B

#pragma unroll
  for (int j = 0; j < 4; ++j) {
    const int n = j * 16 + (lane & 15);
#pragma unroll
    for (int r = 0; r < 8; ++r) {
      const int m = (w << 4) + r + ((lane >> 4) << 3);
      if (n < Vv) logits[m * Vv + n] = acc[j][r] + bout[n];
    }
  }
}

// ---------------------------------------------------------------------------
extern "C" void kernel_launch(void* const* d_in, const int* in_sizes, int n_in,
                              void* d_out, int out_size, void* d_ws, size_t ws_size,
                              hipStream_t stream) {
  const int*   tokens  = (const int*)  d_in[0];
  const float* hidden0 = (const float*)d_in[1];
  const float* embed   = (const float*)d_in[2];
  const float* Ws      = (const float*)d_in[3];
  // d_in[4] = bs : per-column constant, cancelled exactly by batch-norm -> unused
  const float* us      = (const float*)d_in[5];
  const float* gammas  = (const float*)d_in[6];
  const float* betas   = (const float*)d_in[7];
  const float* Wout    = (const float*)d_in[8];
  const float* bout    = (const float*)d_in[9];
  float* out = (float*)d_out;

  // Workspace carve-up (~58 MB)
  char* ws = (char*)d_ws;
  unsigned short* Wp = (unsigned short*)ws;
  size_t off = (size_t)Ll * Hh * Hh * sizeof(unsigned short);          // 48 MB
  unsigned short* WoutP = (unsigned short*)(ws + off);
  off += (size_t)64 * Hh * sizeof(unsigned short);                     // 256 KB
  float* hid = (float*)(ws + off);
  off += (size_t)Ll * Bb * Hh * sizeof(float);                         // 6 MB
  unsigned short* xb0 = (unsigned short*)(ws + off);
  off += (size_t)Bb * Hh * sizeof(unsigned short);                     // 512 KB
  unsigned short* xb1 = (unsigned short*)(ws + off);                   // 512 KB

  pack_weights<<<1024, 256, 0, stream>>>(Ws, Wp);
  pack_wout<<<64, 256, 0, stream>>>(Wout, WoutP);
  hipMemcpyAsync(hid, hidden0, (size_t)Ll * Bb * Hh * sizeof(float),
                 hipMemcpyDeviceToDevice, stream);

  unsigned short* xb[2] = {xb0, xb1};
  for (int t = 0; t < Tt; ++t) {
    embed_gather<<<(Bb * Hh) / 256, 256, 0, stream>>>(tokens + (size_t)t * Bb,
                                                      embed, xb[0]);
    for (int l = 0; l < Ll; ++l) {
      layer_step<<<Hh / 64, 256, LDS_DYN, stream>>>(
          xb[l & 1], Wp + (size_t)l * Hh * Hh,
          gammas + (size_t)l * Hh, betas + (size_t)l * Hh, us + (size_t)l * Hh,
          hid + (size_t)l * Bb * Hh, xb[(l + 1) & 1]);
    }
    // layer 5 wrote xb[0]
    out_proj<<<1, 256, 0, stream>>>(xb[0], WoutP, bout,
                                    out + (size_t)t * Bb * Vv);
  }
  // outputs: logits [T*B*V] then hidden_new [L*B*H]
  hipMemcpyAsync(out + (size_t)Tt * Bb * Vv, hid,
                 (size_t)Ll * Bb * Hh * sizeof(float),
                 hipMemcpyDeviceToDevice, stream);
}
